// Token_indicator_OTE2TS3_89790586290590
// MI455X (gfx1250) — compile-verified, matrix operand order checked by reading, not run on hardware
//
#include <hip/hip_runtime.h>
#include <hip/hip_bf16.h>
#include <math.h>

#define HDIM   768
#define NLAYER 12
#define NHEAD  12
#define DHEAD  64
#define FFDIM  3072
#define DGRU   60
#define SEQ    100
#define BATCH  32
#define TOK    (BATCH * SEQ)

typedef __attribute__((ext_vector_type(16))) __bf16 v16bf;
typedef __attribute__((ext_vector_type(8)))  __bf16 v8bf;
typedef __attribute__((ext_vector_type(8)))  float  v8f;

// K-pair base within a bf16 WMMA fragment VGPR v (ISA 7.12.2, 16-bit A 16x32)
__device__ __forceinline__ int frag_kbase(int v, int half) {
    return (v < 4 ? 2 * v : 16 + 2 * (v - 4)) + 8 * half;
}

// ---------------------------------------------------------------------------
// Tiled bf16 WMMA GEMM: C[M,N] = act( A[M,K] @ B + bias )
// B logical (K,N): BT==0 -> stored K x N row-major ; BT==1 -> N x K
// ACT: 0 = none, 1 = exact GELU
// Block tile 64x128, 8 waves, 4 WMMA accumulators per wave, reg double-buffer,
// global_prefetch two K-tiles ahead.
// ---------------------------------------------------------------------------
#define BM  64
#define BN  128
#define BKK 32
#define LDK 40   // LDS row stride in bf16 (80B: keeps 16B alignment, shifts banks)

template <int BT, int ACT>
__global__ __launch_bounds__(256)
void gemm_wmma(const float* __restrict__ A, const float* __restrict__ B,
               const float* __restrict__ bias, float* __restrict__ C,
               int M, int N, int K)
{
    __shared__ __bf16 As[BM][LDK];
    __shared__ __bf16 Bs[BN][LDK];

    const int tid  = threadIdx.x;
    const int lane = tid & 31;
    const int wid  = tid >> 5;
    const int wm   = wid & 3;   // 16-row slice
    const int wn   = wid >> 2;  // 64-col half
    const int row0 = blockIdx.y * BM;
    const int col0 = blockIdx.x * BN;
    const int half = lane >> 4;
    const int idx  = lane & 15;

    // A staging map: 64 rows x 32 k, 8 floats/thread
    const int ar = tid >> 2;
    const int ac = (tid & 3) * 8;
    // B staging map (BT==1): 128 n-rows x 32 k, 16 floats/thread
    const int bt_n = tid >> 1;
    const int bt_c = (tid & 1) * 16;
    // B staging map (BT==0): 32 k-rows x 128 n, 16 floats/thread
    const int bn_k = tid >> 3;
    const int bn_c = (tid & 7) * 16;

    float aReg[8];
    float bReg[16];

    auto loadA = [&](int k0) {
        const int gr = row0 + ar;
        const int gk = k0 + ac;
        if (gr < M && gk + 8 <= K) {
            const float4 x = *(const float4*)(A + (size_t)gr * K + gk);
            const float4 y = *(const float4*)(A + (size_t)gr * K + gk + 4);
            aReg[0] = x.x; aReg[1] = x.y; aReg[2] = x.z; aReg[3] = x.w;
            aReg[4] = y.x; aReg[5] = y.y; aReg[6] = y.z; aReg[7] = y.w;
        } else {
#pragma unroll
            for (int j = 0; j < 8; ++j)
                aReg[j] = (gr < M && gk + j < K) ? A[(size_t)gr * K + gk + j] : 0.0f;
        }
    };

    auto loadB = [&](int k0) {
        if (BT) {
            const int gc = col0 + bt_n;
            const int gk = k0 + bt_c;
            if (gc < N && gk + 16 <= K) {
                const float* p = B + (size_t)gc * K + gk;
#pragma unroll
                for (int q4 = 0; q4 < 4; ++q4) {
                    const float4 x = *(const float4*)(p + 4 * q4);
                    bReg[4 * q4 + 0] = x.x; bReg[4 * q4 + 1] = x.y;
                    bReg[4 * q4 + 2] = x.z; bReg[4 * q4 + 3] = x.w;
                }
            } else {
#pragma unroll
                for (int j = 0; j < 16; ++j)
                    bReg[j] = (gc < N && gk + j < K) ? B[(size_t)gc * K + gk + j] : 0.0f;
            }
        } else {
            const int gk = k0 + bn_k;
            const int gc = col0 + bn_c;
            if (gk < K && gc + 16 <= N) {
                const float* p = B + (size_t)gk * N + gc;
#pragma unroll
                for (int q4 = 0; q4 < 4; ++q4) {
                    const float4 x = *(const float4*)(p + 4 * q4);
                    bReg[4 * q4 + 0] = x.x; bReg[4 * q4 + 1] = x.y;
                    bReg[4 * q4 + 2] = x.z; bReg[4 * q4 + 3] = x.w;
                }
            } else {
#pragma unroll
                for (int j = 0; j < 16; ++j)
                    bReg[j] = (gk < K && gc + j < N) ? B[(size_t)gk * N + gc + j] : 0.0f;
            }
        }
    };

    auto storeTiles = [&]() {
        {
            v8bf t;
#pragma unroll
            for (int j = 0; j < 8; ++j) t[j] = (__bf16)aReg[j];
            *(v8bf*)&As[ar][ac] = t;
        }
        if (BT) {
            v8bf t0, t1;
#pragma unroll
            for (int j = 0; j < 8; ++j) { t0[j] = (__bf16)bReg[j]; t1[j] = (__bf16)bReg[8 + j]; }
            *(v8bf*)&Bs[bt_n][bt_c]     = t0;
            *(v8bf*)&Bs[bt_n][bt_c + 8] = t1;
        } else {
#pragma unroll
            for (int j = 0; j < 16; ++j) Bs[bn_c + j][bn_k] = (__bf16)bReg[j];
        }
    };

    v8f acc[4];
#pragma unroll
    for (int nt = 0; nt < 4; ++nt) acc[nt] = (v8f){};

    loadA(0);
    loadB(0);
    for (int k0 = 0; k0 < K; k0 += BKK) {
        storeTiles();
        __syncthreads();
        if (k0 + BKK < K) {
            loadA(k0 + BKK);
            loadB(k0 + BKK);
        }
        // pull the tile after next toward the caches (global_prefetch_b8)
        if (k0 + 2 * BKK < K) {
            __builtin_prefetch(A + (size_t)(row0 + ar) * K + k0 + 2 * BKK + ac, 0, 1);
            if (BT)
                __builtin_prefetch(B + (size_t)(col0 + bt_n) * K + k0 + 2 * BKK + bt_c, 0, 1);
            else
                __builtin_prefetch(B + (size_t)(k0 + 2 * BKK + bn_k) * N + col0 + bn_c, 0, 1);
        }

        v16bf af;
        const __bf16* arow = &As[wm * 16 + idx][0];
#pragma unroll
        for (int v = 0; v < 8; ++v) {
            const int kb = frag_kbase(v, half);
            af[2 * v]     = arow[kb];
            af[2 * v + 1] = arow[kb + 1];
        }
#pragma unroll
        for (int nt = 0; nt < 4; ++nt) {
            v16bf bf;
            const __bf16* brow = &Bs[wn * 64 + nt * 16 + idx][0];
#pragma unroll
            for (int v = 0; v < 8; ++v) {
                const int kb = frag_kbase(v, half);
                bf[2 * v]     = brow[kb];
                bf[2 * v + 1] = brow[kb + 1];
            }
            acc[nt] = __builtin_amdgcn_wmma_f32_16x16x32_bf16(false, af, false, bf,
                                                              (short)0, acc[nt], false, false);
        }
        __syncthreads();
    }

    const bool full = (row0 + BM <= M) && (col0 + BN <= N);
#pragma unroll
    for (int nt = 0; nt < 4; ++nt) {
        const int col = col0 + wn * 64 + nt * 16 + idx;
        if (full) {
            const float bv = bias ? bias[col] : 0.0f;
#pragma unroll
            for (int r = 0; r < 8; ++r) {
                const int row = row0 + wm * 16 + r + 8 * half;
                float v = acc[nt][r] + bv;
                if (ACT == 1) v = 0.5f * v * (1.0f + erff(v * 0.70710678118654752f));
                C[(size_t)row * N + col] = v;
            }
        } else if (col < N) {
            const float bv = bias ? bias[col] : 0.0f;
#pragma unroll
            for (int r = 0; r < 8; ++r) {
                const int row = row0 + wm * 16 + r + 8 * half;
                if (row < M) {
                    float v = acc[nt][r] + bv;
                    if (ACT == 1) v = 0.5f * v * (1.0f + erff(v * 0.70710678118654752f));
                    C[(size_t)row * N + col] = v;
                }
            }
        }
    }
}

// ---------------------------------------------------------------------------
// Attention scores: scores[b,h,q,k] = (q . k)/8 + mask_bias   (one wave/tile)
// Out-of-range rows/cols are clamped: their outputs are never stored.
// ---------------------------------------------------------------------------
__global__ __launch_bounds__(32)
void attn_scores(const float* __restrict__ q, const float* __restrict__ k,
                 const int* __restrict__ amask, float* __restrict__ sc)
{
    const int bh = blockIdx.z;
    const int b  = bh / NHEAD;
    const int hh = bh % NHEAD;
    const int qt = blockIdx.y;
    const int kt = blockIdx.x;
    const int lane = threadIdx.x & 31;
    const int half = lane >> 4;
    const int idx  = lane & 15;

    const int qi  = qt * 16 + idx;
    const int ki  = kt * 16 + idx;
    const int qic = qi < SEQ ? qi : SEQ - 1;
    const int kic = ki < SEQ ? ki : SEQ - 1;
    const float* qp = q + ((size_t)(b * SEQ + qic)) * HDIM + hh * DHEAD;
    const float* kp = k + ((size_t)(b * SEQ + kic)) * HDIM + hh * DHEAD;

    v8f acc = {};
#pragma unroll
    for (int ks = 0; ks < 2; ++ks) {  // DHEAD = 64 = 2 x 32
        v16bf af, bf;
#pragma unroll
        for (int v = 0; v < 8; ++v) {
            const int kb = frag_kbase(v, half) + ks * 32;
            const float2 qa = *(const float2*)(qp + kb);
            const float2 ka = *(const float2*)(kp + kb);
            af[2 * v] = (__bf16)qa.x; af[2 * v + 1] = (__bf16)qa.y;
            bf[2 * v] = (__bf16)ka.x; bf[2 * v + 1] = (__bf16)ka.y;
        }
        acc = __builtin_amdgcn_wmma_f32_16x16x32_bf16(false, af, false, bf,
                                                      (short)0, acc, false, false);
    }

    // hoisted mask bias: depends only on the key column (clamped load, no branch)
    const int  kcol = kt * 16 + idx;
    const int  kcc  = kcol < SEQ ? kcol : SEQ - 1;
    const float mb  = (1.0f - (float)amask[b * SEQ + kcc]) * -10000.0f;
    float* op = sc + ((size_t)bh * SEQ + (size_t)qt * 16) * SEQ + kcol;

    if (kcol < SEQ) {
        if ((qt + 1) * 16 <= SEQ) {
            // full row tile: unconditional stores
#pragma unroll
            for (int r = 0; r < 8; ++r) {
                const int rr = r + 8 * half;
                op[(size_t)rr * SEQ] = acc[r] * 0.125f + mb;
            }
        } else {
#pragma unroll
            for (int r = 0; r < 8; ++r) {
                const int rr = r + 8 * half;
                if (qt * 16 + rr < SEQ)
                    op[(size_t)rr * SEQ] = acc[r] * 0.125f + mb;
            }
        }
    }
}

// ---------------------------------------------------------------------------
// Row softmax over SEQ keys; 8 waves per block, one row per wave
// ---------------------------------------------------------------------------
__global__ __launch_bounds__(256)
void softmax_rows(float* __restrict__ sc, int nrows)
{
    const int row  = blockIdx.x * 8 + (threadIdx.x >> 5);
    const int lane = threadIdx.x & 31;
    if (row >= nrows) return;
    const size_t base = (size_t)row * SEQ;

    float mx = -1e30f;
    for (int i = lane; i < SEQ; i += 32) mx = fmaxf(mx, sc[base + i]);
#pragma unroll
    for (int o = 16; o > 0; o >>= 1) mx = fmaxf(mx, __shfl_xor(mx, o, 32));
    float s = 0.0f;
    for (int i = lane; i < SEQ; i += 32) {
        const float e = expf(sc[base + i] - mx);
        sc[base + i] = e;
        s += e;
    }
#pragma unroll
    for (int o = 16; o > 0; o >>= 1) s += __shfl_xor(s, o, 32);
    const float inv = 1.0f / s;
    for (int i = lane; i < SEQ; i += 32) sc[base + i] *= inv;
}

// ---------------------------------------------------------------------------
// ctx[b,q,h,:] = softmax(scores) @ v      (one wave per 16x16 tile)
// K (= key dim) padded 100 -> 128: A-side zero-selected, V side times zero.
// ---------------------------------------------------------------------------
__global__ __launch_bounds__(32)
void attn_ctx(const float* __restrict__ sc, const float* __restrict__ vv,
              float* __restrict__ ctx)
{
    const int bh = blockIdx.z;
    const int b  = bh / NHEAD;
    const int hh = bh % NHEAD;
    const int qt = blockIdx.y;   // 0..6
    const int nt = blockIdx.x;   // 0..3 (DHEAD/16)
    const int lane = threadIdx.x & 31;
    const int half = lane >> 4;
    const int idx  = lane & 15;

    const int qi  = qt * 16 + idx;
    const int qic = qi < SEQ ? qi : SEQ - 1;
    const int n   = nt * 16 + idx;
    const float* ap = sc + ((size_t)bh * SEQ + qic) * SEQ;

    v8f acc = {};
    for (int k0 = 0; k0 < 128; k0 += 32) {
        v16bf af, bf;
#pragma unroll
        for (int v = 0; v < 8; ++v) {
            const int kk  = k0 + frag_kbase(v, half);     // even; pair in/out together
            const bool in = kk < SEQ;
            const int  kc = in ? kk : 0;
            const float2 av = *(const float2*)(ap + kc);
            const int kc1 = in ? kk + 1 : 0;
            const float b0 = vv[((size_t)(b * SEQ + kc)) * HDIM + hh * DHEAD + n];
            const float b1 = vv[((size_t)(b * SEQ + kc1)) * HDIM + hh * DHEAD + n];
            af[2 * v]     = (__bf16)(in ? av.x : 0.0f);
            af[2 * v + 1] = (__bf16)(in ? av.y : 0.0f);
            bf[2 * v]     = (__bf16)b0;
            bf[2 * v + 1] = (__bf16)b1;
        }
        acc = __builtin_amdgcn_wmma_f32_16x16x32_bf16(false, af, false, bf,
                                                      (short)0, acc, false, false);
    }

    float* op = ctx + ((size_t)(b * SEQ + qt * 16)) * HDIM + hh * DHEAD + n;
    if ((qt + 1) * 16 <= SEQ) {
        // full row tile: unconditional stores
#pragma unroll
        for (int r = 0; r < 8; ++r) {
            const int rr = r + 8 * half;
            op[(size_t)rr * HDIM] = acc[r];
        }
    } else {
#pragma unroll
        for (int r = 0; r < 8; ++r) {
            const int rr = r + 8 * half;
            if (qt * 16 + rr < SEQ)
                op[(size_t)rr * HDIM] = acc[r];
        }
    }
}

// ---------------------------------------------------------------------------
// h[t] = LayerNorm(h[t] (+ delta[t])) * g + b   — block per token, HDIM=768
// ---------------------------------------------------------------------------
__global__ __launch_bounds__(256)
void add_layernorm(float* __restrict__ h, const float* __restrict__ delta,
                   const float* __restrict__ g, const float* __restrict__ bb)
{
    __shared__ float red[9];
    const int t   = blockIdx.x;
    const int tid = threadIdx.x;

    float x[3];
    float s = 0.0f;
#pragma unroll
    for (int j = 0; j < 3; ++j) {
        const int i = tid + j * 256;
        float v = h[(size_t)t * HDIM + i];
        if (delta) v += delta[(size_t)t * HDIM + i];
        x[j] = v;
        s += v;
    }
#pragma unroll
    for (int o = 16; o > 0; o >>= 1) s += __shfl_xor(s, o, 32);
    if ((tid & 31) == 0) red[tid >> 5] = s;
    __syncthreads();
    if (tid == 0) { float a = 0; for (int i = 0; i < 8; ++i) a += red[i]; red[8] = a; }
    __syncthreads();
    const float mean = red[8] * (1.0f / HDIM);
    __syncthreads();

    float vs = 0.0f;
#pragma unroll
    for (int j = 0; j < 3; ++j) { const float d = x[j] - mean; vs += d * d; }
#pragma unroll
    for (int o = 16; o > 0; o >>= 1) vs += __shfl_xor(vs, o, 32);
    if ((tid & 31) == 0) red[tid >> 5] = vs;
    __syncthreads();
    if (tid == 0) { float a = 0; for (int i = 0; i < 8; ++i) a += red[i]; red[8] = a; }
    __syncthreads();
    const float var  = red[8] * (1.0f / HDIM);
    const float rstd = rsqrtf(var + 1e-12f);
#pragma unroll
    for (int j = 0; j < 3; ++j) {
        const int i = tid + j * 256;
        h[(size_t)t * HDIM + i] = (x[j] - mean) * rstd * g[i] + bb[i];
    }
}

// ---------------------------------------------------------------------------
// Embedding gather
// ---------------------------------------------------------------------------
__global__ __launch_bounds__(256)
void embed(const int* __restrict__ ids, const int* __restrict__ tt,
           const float* __restrict__ we, const float* __restrict__ pe,
           const float* __restrict__ te, float* __restrict__ h)
{
    const int t  = blockIdx.x;
    const int s  = t % SEQ;
    const int id = ids[t];
    const int ty = tt[t];
    for (int i = threadIdx.x; i < HDIM; i += 256)
        h[(size_t)t * HDIM + i] =
            we[(size_t)id * HDIM + i] + pe[(size_t)s * HDIM + i] + te[(size_t)ty * HDIM + i];
}

// ---------------------------------------------------------------------------
// GRU recurrence: grid (BATCH, 2 dirs), gi precomputed = x@Wih^T + bih
// ---------------------------------------------------------------------------
__global__ __launch_bounds__(192)
void gru_rec(const float* __restrict__ gi, const float* __restrict__ whh,
             const float* __restrict__ bhh, float* __restrict__ out)
{
    const int b   = blockIdx.x;
    const int dir = blockIdx.y;
    const float* W  = whh + (size_t)dir * 3 * DGRU * DGRU;  // (180,60)
    const float* bh = bhh + (size_t)dir * 3 * DGRU;
    __shared__ float hcur[DGRU];
    __shared__ float gh[3 * DGRU];
    const int tid = threadIdx.x;
    if (tid < DGRU) hcur[tid] = 0.0f;
    __syncthreads();

    for (int step = 0; step < SEQ; ++step) {
        const int t = dir ? (SEQ - 1 - step) : step;
        const float* gt = gi + ((size_t)dir * TOK + (size_t)b * SEQ + t) * (3 * DGRU);
        // prefetch next timestep's input-gate row
        if (step + 1 < SEQ) {
            const int tn = dir ? (SEQ - 2 - step) : step + 1;
            __builtin_prefetch(gi + ((size_t)dir * TOK + (size_t)b * SEQ + tn) * (3 * DGRU) + tid, 0, 1);
        }
        if (tid < 3 * DGRU) {
            float s = bh[tid];
            const float* wr = W + (size_t)tid * DGRU;
            for (int k = 0; k < DGRU; ++k) s += hcur[k] * wr[k];
            gh[tid] = s;
        }
        __syncthreads();
        float h2 = 0.0f;
        if (tid < DGRU) {
            const float ir  = gt[tid];
            const float iz  = gt[DGRU + tid];
            const float inn = gt[2 * DGRU + tid];
            const float r = 1.0f / (1.0f + expf(-(ir + gh[tid])));
            const float z = 1.0f / (1.0f + expf(-(iz + gh[DGRU + tid])));
            const float n = tanhf(inn + r * gh[2 * DGRU + tid]);
            h2 = (1.0f - z) * n + z * hcur[tid];
            out[((size_t)b * SEQ + t) * (2 * DGRU) + dir * DGRU + tid] = h2;
        }
        __syncthreads();
        if (tid < DGRU) hcur[tid] = h2;
        __syncthreads();
    }
}

// ---------------------------------------------------------------------------
// Gated blend scan over time, block per batch element
// ---------------------------------------------------------------------------
__global__ __launch_bounds__(128)
void gate_blend(const float* __restrict__ ts_hs, const float* __restrict__ Wg,
                float* __restrict__ ts_tilde)
{
    const int b   = blockIdx.x;
    const int tid = threadIdx.x;
    __shared__ float hprev[2 * DGRU];
    __shared__ float xs[2 * DGRU];
    if (tid < 2 * DGRU) {
        const float v = ts_hs[((size_t)b * SEQ) * (2 * DGRU) + tid];
        hprev[tid] = v;
        ts_tilde[((size_t)b * SEQ) * (2 * DGRU) + tid] = v;
    }
    __syncthreads();
    for (int t = 1; t < SEQ; ++t) {
        if (tid < 2 * DGRU) xs[tid] = ts_hs[((size_t)b * SEQ + t) * (2 * DGRU) + tid];
        __syncthreads();
        float hnew = 0.0f;
        if (tid < 2 * DGRU) {
            float s = 0.0f;
            for (int k = 0; k < 2 * DGRU; ++k) s += xs[k] * Wg[k * (2 * DGRU) + tid];
            const float g = 1.0f / (1.0f + expf(-s));
            hnew = g * xs[tid] + (1.0f - g) * hprev[tid];
            ts_tilde[((size_t)b * SEQ + t) * (2 * DGRU) + tid] = hnew;
        }
        __syncthreads();
        if (tid < 2 * DGRU) hprev[tid] = hnew;
        __syncthreads();
    }
}

// ---------------------------------------------------------------------------
// Heads: p_ote logits, p_ts_tilde emissions (thread per token)
// ---------------------------------------------------------------------------
__global__ __launch_bounds__(128)
void head_mix(const float* __restrict__ lm_hs, const float* __restrict__ lm_ts,
              const float* __restrict__ Wote, const float* __restrict__ bote,
              const float* __restrict__ Wts, const float* __restrict__ bts,
              float* __restrict__ p_ote, float* __restrict__ p_ts_tilde)
{
    const int t = blockIdx.x * blockDim.x + threadIdx.x;
    if (t >= TOK) return;

    float po[4], pt[7];
    const float* x = lm_hs + (size_t)t * (2 * DGRU);
    for (int j = 0; j < 4; ++j) {
        float s = bote[j];
        for (int k = 0; k < 2 * DGRU; ++k) s += x[k] * Wote[k * 4 + j];
        po[j] = s;
    }
    const float* y = lm_ts + (size_t)t * (2 * DGRU);
    for (int j = 0; j < 7; ++j) {
        float s = bts[j];
        for (int k = 0; k < 2 * DGRU; ++k) s += y[k] * Wts[k * 7 + j];
        pt[j] = s;
    }

    float m = po[0];
    for (int j = 1; j < 4; ++j) m = fmaxf(m, po[j]);
    float posm[4], se = 0.0f;
    for (int j = 0; j < 4; ++j) { posm[j] = expf(po[j] - m); se += posm[j]; }
    float alpha = 0.0f;
    for (int j = 0; j < 4; ++j) { posm[j] /= se; alpha += posm[j] * posm[j]; }
    alpha = (1.0f - alpha) * 0.5f;

    m = pt[0];
    for (int j = 1; j < 7; ++j) m = fmaxf(m, pt[j]);
    float ptsm[7]; se = 0.0f;
    for (int j = 0; j < 7; ++j) { ptsm[j] = expf(pt[j] - m); se += ptsm[j]; }
    for (int j = 0; j < 7; ++j) ptsm[j] /= se;

    const float TS[4][7] = {
        {1.0f, 0, 0, 0, 0, 0, 0},
        {0, 0.5f, 0, 0, 0.5f, 0, 0},
        {0, 0, 0.5f, 0, 0, 0.5f, 0},
        {0, 0, 0, 0.5f, 0, 0, 0.5f}};
    for (int j = 0; j < 7; ++j) {
        float o2t = 0.0f;
        for (int i = 0; i < 4; ++i) o2t += po[i] * TS[i][j];
        p_ts_tilde[(size_t)t * 7 + j] = alpha * o2t + (1.0f - alpha) * ptsm[j];
    }
    for (int j = 0; j < 4; ++j) p_ote[(size_t)t * 4 + j] = po[j];
}

// ---------------------------------------------------------------------------
// CRF NLL per batch element; writes (den - num) into partial[slot + b]
// ---------------------------------------------------------------------------
__global__ __launch_bounds__(32)
void crf_nll(const float* __restrict__ em, const int* __restrict__ tags,
             const int* __restrict__ amask, const float* __restrict__ start,
             const float* __restrict__ endv, const float* __restrict__ trans,
             int nstate, float* __restrict__ partial, int slot)
{
    const int b = threadIdx.x;
    if (b >= BATCH) return;
    const float* E  = em + (size_t)b * SEQ * nstate;
    const int*   tg = tags + b * SEQ;

    float num = start[tg[0]] + E[tg[0]];
    int cnt = amask[b * SEQ + 0];
    for (int t = 1; t < SEQ; ++t) {
        const float m = (float)amask[b * SEQ + t];
        cnt += amask[b * SEQ + t];
        num += (trans[tg[t - 1] * nstate + tg[t]] + E[(size_t)t * nstate + tg[t]]) * m;
    }
    num += endv[tg[cnt - 1]];

    float alpha[7];
    for (int j = 0; j < nstate; ++j) alpha[j] = start[j] + E[j];
    for (int t = 1; t < SEQ; ++t) {
        const float m = (float)amask[b * SEQ + t];
        float nxt[7];
        for (int j = 0; j < nstate; ++j) {
            float mx = -1e30f;
            for (int i = 0; i < nstate; ++i) mx = fmaxf(mx, alpha[i] + trans[i * nstate + j]);
            float s = 0.0f;
            for (int i = 0; i < nstate; ++i) s += expf(alpha[i] + trans[i * nstate + j] - mx);
            nxt[j] = mx + logf(s) + E[(size_t)t * nstate + j];
        }
        for (int j = 0; j < nstate; ++j) alpha[j] = (m > 0.0f) ? nxt[j] : alpha[j];
    }
    float mx = -1e30f;
    for (int j = 0; j < nstate; ++j) mx = fmaxf(mx, alpha[j] + endv[j]);
    float s = 0.0f;
    for (int j = 0; j < nstate; ++j) s += expf(alpha[j] + endv[j] - mx);
    const float den = mx + logf(s);

    partial[slot + b] = den - num;
}

__global__ void finalize(const float* __restrict__ partial, float* __restrict__ out)
{
    if (threadIdx.x == 0 && blockIdx.x == 0) {
        float s = 0.0f;
        for (int i = 0; i < 2 * BATCH; ++i) s += partial[i];
        out[0] = s;
    }
}

// ---------------------------------------------------------------------------
extern "C" void kernel_launch(void* const* d_in, const int* in_sizes, int n_in,
                              void* d_out, int out_size, void* d_ws, size_t ws_size,
                              hipStream_t stream)
{
    (void)in_sizes; (void)n_in; (void)out_size; (void)ws_size;

    const int*   input_ids  = (const int*)d_in[0];
    const int*   amask      = (const int*)d_in[1];
    const int*   ttids      = (const int*)d_in[2];
    const int*   ote_labels = (const int*)d_in[3];
    const int*   ts_labels  = (const int*)d_in[4];
    const float* word_emb   = (const float*)d_in[5];
    const float* pos_emb    = (const float*)d_in[6];
    const float* type_emb   = (const float*)d_in[7];
    const float* emb_g      = (const float*)d_in[8];
    const float* emb_b      = (const float*)d_in[9];
    const float* Wq   = (const float*)d_in[10];
    const float* bq   = (const float*)d_in[11];
    const float* Wk   = (const float*)d_in[12];
    const float* bk   = (const float*)d_in[13];
    const float* Wv   = (const float*)d_in[14];
    const float* bv   = (const float*)d_in[15];
    const float* Wo   = (const float*)d_in[16];
    const float* bo   = (const float*)d_in[17];
    const float* ln1g = (const float*)d_in[18];
    const float* ln1b = (const float*)d_in[19];
    const float* W1   = (const float*)d_in[20];
    const float* bf1  = (const float*)d_in[21];
    const float* W2   = (const float*)d_in[22];
    const float* bf2  = (const float*)d_in[23];
    const float* ln2g = (const float*)d_in[24];
    const float* ln2b = (const float*)d_in[25];
    const float* g1_wih = (const float*)d_in[26];
    const float* g1_whh = (const float*)d_in[27];
    const float* g1_bih = (const float*)d_in[28];
    const float* g1_bhh = (const float*)d_in[29];
    const float* g2_wih = (const float*)d_in[30];
    const float* g2_whh = (const float*)d_in[31];
    const float* g2_bih = (const float*)d_in[32];
    const float* g2_bhh = (const float*)d_in[33];
    const float* stm_lm_W = (const float*)d_in[34];
    const float* stm_lm_b = (const float*)d_in[35];
    const float* stm_ts_W = (const float*)d_in[36];
    const float* stm_ts_b = (const float*)d_in[37];
    const float* W_gate   = (const float*)d_in[38];
    const float* fc_ote_W = (const float*)d_in[39];
    const float* fc_ote_b = (const float*)d_in[40];
    const float* fc_ts_W  = (const float*)d_in[41];
    const float* fc_ts_b  = (const float*)d_in[42];
    const float* ote_start = (const float*)d_in[43];
    const float* ote_end   = (const float*)d_in[44];
    const float* ote_trans = (const float*)d_in[45];
    const float* ts_start  = (const float*)d_in[46];
    const float* ts_end    = (const float*)d_in[47];
    const float* ts_trans  = (const float*)d_in[48];
    float* out = (float*)d_out;

    // workspace partition (floats)
    float* ws = (float*)d_ws;
    size_t off = 0;
    float* h        = ws + off; off += (size_t)TOK * HDIM;
    float* qb       = ws + off; off += (size_t)TOK * HDIM;   // also ctx
    float* kb2      = ws + off; off += (size_t)TOK * HDIM;
    float* vb2      = ws + off; off += (size_t)TOK * HDIM;
    float* tmp      = ws + off; off += (size_t)TOK * HDIM;
    float* sc       = ws + off; off += (size_t)BATCH * NHEAD * SEQ * SEQ;
    float* ffn      = ws + off; off += (size_t)TOK * FFDIM;
    float* gi       = ws + off; off += (size_t)2 * TOK * 3 * DGRU;
    float* ote_hs   = ws + off; off += (size_t)TOK * 2 * DGRU;
    float* ts_hs    = ws + off; off += (size_t)TOK * 2 * DGRU;
    float* ts_tilde = ws + off; off += (size_t)TOK * 2 * DGRU;
    float* lm_hs    = ws + off; off += (size_t)TOK * 2 * DGRU;
    float* lm_ts    = ws + off; off += (size_t)TOK * 2 * DGRU;
    float* p_ote    = ws + off; off += (size_t)TOK * 4;
    float* p_tst    = ws + off; off += (size_t)TOK * 7;
    float* partial  = ws + off; off += 64;

    const dim3 gH((HDIM + BN - 1) / BN, (TOK + BM - 1) / BM);
    const dim3 gF((FFDIM + BN - 1) / BN, (TOK + BM - 1) / BM);
    const dim3 g180((3 * DGRU + BN - 1) / BN, (TOK + BM - 1) / BM);
    const dim3 g120((2 * DGRU + BN - 1) / BN, (TOK + BM - 1) / BM);
    const int QT = (SEQ + 15) / 16;  // 7
    const int NROWS = BATCH * NHEAD * SEQ;

    // embeddings + LN
    embed<<<TOK, 256, 0, stream>>>(input_ids, ttids, word_emb, pos_emb, type_emb, h);
    add_layernorm<<<TOK, 256, 0, stream>>>(h, nullptr, emb_g, emb_b);

    // BERT encoder
    for (int l = 0; l < NLAYER; ++l) {
        const size_t oHH = (size_t)l * HDIM * HDIM;
        const size_t oH  = (size_t)l * HDIM;
        const size_t oHF = (size_t)l * HDIM * FFDIM;
        const size_t oF  = (size_t)l * FFDIM;

        gemm_wmma<0, 0><<<gH, 256, 0, stream>>>(h, Wq + oHH, bq + oH, qb, TOK, HDIM, HDIM);
        gemm_wmma<0, 0><<<gH, 256, 0, stream>>>(h, Wk + oHH, bk + oH, kb2, TOK, HDIM, HDIM);
        gemm_wmma<0, 0><<<gH, 256, 0, stream>>>(h, Wv + oHH, bv + oH, vb2, TOK, HDIM, HDIM);

        attn_scores<<<dim3(QT, QT, BATCH * NHEAD), 32, 0, stream>>>(qb, kb2, amask, sc);
        softmax_rows<<<(NROWS + 7) / 8, 256, 0, stream>>>(sc, NROWS);
        attn_ctx<<<dim3(DHEAD / 16, QT, BATCH * NHEAD), 32, 0, stream>>>(sc, vb2, qb);

        gemm_wmma<0, 0><<<gH, 256, 0, stream>>>(qb, Wo + oHH, bo + oH, tmp, TOK, HDIM, HDIM);
        add_layernorm<<<TOK, 256, 0, stream>>>(h, tmp, ln1g + oH, ln1b + oH);

        gemm_wmma<0, 1><<<gF, 256, 0, stream>>>(h, W1 + oHF, bf1 + oF, ffn, TOK, FFDIM, HDIM);
        gemm_wmma<0, 0><<<gH, 256, 0, stream>>>(ffn, W2 + oHF, bf2 + oH, tmp, TOK, HDIM, FFDIM);
        add_layernorm<<<TOK, 256, 0, stream>>>(h, tmp, ln2g + oH, ln2b + oH);
    }

    // BiGRU 1: gi = seq @ Wih^T + bih (per dir), then recurrence
    for (int dir = 0; dir < 2; ++dir)
        gemm_wmma<1, 0><<<g180, 256, 0, stream>>>(h, g1_wih + (size_t)dir * 3 * DGRU * HDIM,
                                                  g1_bih + (size_t)dir * 3 * DGRU,
                                                  gi + (size_t)dir * TOK * 3 * DGRU,
                                                  TOK, 3 * DGRU, HDIM);
    gru_rec<<<dim3(BATCH, 2), 192, 0, stream>>>(gi, g1_whh, g1_bhh, ote_hs);

    gemm_wmma<0, 0><<<g120, 256, 0, stream>>>(ote_hs, stm_lm_W, stm_lm_b, lm_hs,
                                              TOK, 2 * DGRU, 2 * DGRU);

    // BiGRU 2
    for (int dir = 0; dir < 2; ++dir)
        gemm_wmma<1, 0><<<g180, 256, 0, stream>>>(ote_hs,
                                                  g2_wih + (size_t)dir * 3 * DGRU * 2 * DGRU,
                                                  g2_bih + (size_t)dir * 3 * DGRU,
                                                  gi + (size_t)dir * TOK * 3 * DGRU,
                                                  TOK, 3 * DGRU, 2 * DGRU);
    gru_rec<<<dim3(BATCH, 2), 192, 0, stream>>>(gi, g2_whh, g2_bhh, ts_hs);

    gate_blend<<<BATCH, 128, 0, stream>>>(ts_hs, W_gate, ts_tilde);

    gemm_wmma<0, 0><<<g120, 256, 0, stream>>>(ts_tilde, stm_ts_W, stm_ts_b, lm_ts,
                                              TOK, 2 * DGRU, 2 * DGRU);

    head_mix<<<(TOK + 127) / 128, 128, 0, stream>>>(lm_hs, lm_ts, fc_ote_W, fc_ote_b,
                                                    fc_ts_W, fc_ts_b, p_ote, p_tst);

    crf_nll<<<1, 32, 0, stream>>>(p_ote, ote_labels, amask, ote_start, ote_end, ote_trans,
                                  4, partial, 0);
    crf_nll<<<1, 32, 0, stream>>>(p_tst, ts_labels, amask, ts_start, ts_end, ts_trans,
                                  7, partial, 32);
    finalize<<<1, 32, 0, stream>>>(partial, out);
}
